// TemporalAttentionLayer_9844065042897
// MI455X (gfx1250) — matrix operand order repaired and checked by reading, not compiled, not run on hardware
//
#include <hip/hip_runtime.h>
#include <hip/hip_bf16.h>

// ---------------------------------------------------------------------------
// CDNA5 (gfx1250) fused temporal-attention layer.
// All heavy GEMMs use v_wmma_f32_16x16x32_f16 (wave32 WMMA) with double-
// buffered weight panels staged through LDS (shared by all 4 waves) and
// software-pipelined B-fragment loads.
// ---------------------------------------------------------------------------

typedef __attribute__((ext_vector_type(16))) _Float16 v16h;
typedef __attribute__((ext_vector_type(8)))  _Float16 v8h;
typedef __attribute__((ext_vector_type(8)))  float    v8f;

__device__ __forceinline__ v8f zero8() {
  v8f z = {0.f,0.f,0.f,0.f,0.f,0.f,0.f,0.f};
  return z;
}

// A-fragment (16x32 f16, row-major source, row = lane&15).
// ISA 16-bit A layout: lanes 0-15 hold K {0..7, 16..23}; lanes 16-31 hold
// K {8..15, 24..31} (two 16B loads).
__device__ __forceinline__ v16h frag_a(const _Float16* p, int stride, int kbase, int lane) {
  int r  = lane & 15;
  int c0 = kbase + ((lane >> 4) << 3);
  const _Float16* q = p + r * stride + c0;
  v8h lo = *(const v8h*)(q);
  v8h hi = *(const v8h*)(q + 16);
  return __builtin_shufflevector(lo, hi, 0,1,2,3,4,5,6,7,8,9,10,11,12,13,14,15);
}

// B-fragment (32x16 f16). Source stored as [N][K] rows. Lanes 0-15 hold
// K {0..7, 16..23}; lanes 16-31 hold K {8..15, 24..31} (matches A).
__device__ __forceinline__ v16h frag_b(const _Float16* p, int stride, int kbase, int lane) {
  int n  = lane & 15;
  int c0 = kbase + ((lane >> 4) << 3);
  const _Float16* q = p + n * stride + c0;
  v8h lo = *(const v8h*)(q);
  v8h hi = *(const v8h*)(q + 16);
  return __builtin_shufflevector(lo, hi, 0,1,2,3,4,5,6,7,8,9,10,11,12,13,14,15);
}

__device__ __forceinline__ v8f wmma16(v16h a, v16h b, v8f c) {
  return __builtin_amdgcn_wmma_f32_16x16x32_f16(false, a, false, b,
                                                (short)0, c, false, false);
}

// Store a 16x16 f32 C tile as f16. C layout: VGPR r, lanes 0-15 -> M=r,
// N=lane; lanes 16-31 -> M=8+r, N=lane-16.
__device__ __forceinline__ void store_tile_h(_Float16* p, int stride, v8f acc, int lane) {
  int n  = lane & 15;
  int m0 = (lane >> 4) << 3;
#pragma unroll
  for (int r = 0; r < 8; ++r) p[(m0 + r) * stride + n] = (_Float16)acc[r];
}

// Cooperatively stage a 256x32 f16 weight panel (rows of [N][K] weights,
// K-window [kbase, kbase+32)) into LDS. All 8 global_load_b128 issued first
// (clause), then the LDS stores.
__device__ __forceinline__ void load_slab(_Float16* slab, const _Float16* w,
                                          int wstride, int kbase, int tid) {
  v8h tmp[8];
#pragma unroll
  for (int i = 0; i < 8; ++i) {
    int idx = tid + i * 128;
    int rr = idx >> 2, cc = (idx & 3) << 3;
    tmp[i] = *(const v8h*)(w + (long)rr * wstride + kbase + cc);
  }
#pragma unroll
  for (int i = 0; i < 8; ++i) {
    int idx = tid + i * 128;
    int rr = idx >> 2, cc = (idx & 3) << 3;
    *(v8h*)(slab + rr * 32 + cc) = tmp[i];
  }
}

// N=256 GEMM with double-buffered LDS weight panels.  Per K-chunk: issue the
// next panel's copy, then one A-fragment per wave and 16 pipelined WMMAs from
// the current panel into persistent acc[16]; one barrier per chunk.
// A source: aP0 (stride aS0) for kc < split, else aP1 (stride aS1, rebased).
template <int KC>
__device__ __forceinline__ void gemm256(const _Float16* __restrict__ w, int wstride,
                                        const _Float16* aP0, int aS0, int split,
                                        const _Float16* aP1, int aS1,
                                        _Float16* slab /* 2 x 8192 halfs */,
                                        v8f (&acc)[16], int tid, int lane) {
  load_slab(slab, w, wstride, 0, tid);
  __syncthreads();
  for (int kc = 0; kc < KC; ++kc) {
    _Float16* cur = slab + (kc & 1) * 8192;
    if (kc + 1 < KC)
      load_slab(slab + ((kc + 1) & 1) * 8192, w, wstride, (kc + 1) * 32, tid);
    v16h a;
    if (kc < split) a = frag_a(aP0, aS0, kc * 32, lane);
    else            a = frag_a(aP1, aS1, (kc - split) * 32, lane);
    v16h b0 = frag_b(cur, 32, 0, lane);
#pragma unroll
    for (int nt = 0; nt < 16; ++nt) {
      v16h bn = b0;
      if (nt < 15) bn = frag_b(cur + (nt + 1) * 512, 32, 0, lane);
      acc[nt] = wmma16(a, b0, acc[nt]);
      b0 = bn;
    }
    __syncthreads();
  }
}

// ---------------------------------------------------------------------------
// Prep kernels
// ---------------------------------------------------------------------------

__global__ void ta9844_prep_x(const float* __restrict__ x,
                              const float* __restrict__ pos,
                              _Float16* __restrict__ xh, long total) {
  long idx = (long)blockIdx.x * blockDim.x + threadIdx.x;
  if (idx >= total) return;
  int c = (int)(idx & 255);
  int t = (int)((idx >> 8) & 7);
  xh[idx] = (_Float16)(x[idx] + pos[t * 256 + c]);
}

// dst[m][n][k] = f16(src[m][k][n])
__global__ void ta9844_prep_wT(const float* __restrict__ src,
                               _Float16* __restrict__ dst,
                               int M, int K, int N) {
  long total = (long)M * K * N;
  long idx = (long)blockIdx.x * blockDim.x + threadIdx.x;
  if (idx >= total) return;
  long kn  = (long)K * N;
  long m   = idx / kn;
  long rem = idx - m * kn;
  long n   = rem / K;
  long k   = rem - n * K;
  dst[idx] = (_Float16)src[m * kn + k * N + n];
}

// ---------------------------------------------------------------------------
// Shared attention block (tile-local, 64 rows = 8 batches x 8 tokens)
// Qs/Ks/Vs: [64][256] f16 in LDS.  Writes attn-weighted V into dstOuts.
// ---------------------------------------------------------------------------
__device__ __forceinline__ void attention_block(
    const _Float16* Qs, const _Float16* Ks, const _Float16* Vs,
    _Float16* dstOuts, float* ew, _Float16* attnS, int tid, int lane, int wave) {
  // e = Q * K^T / 16, block-diagonal causal mask
  {
    v8f e = zero8();
    for (int kc = 0; kc < 8; ++kc) {
      v16h a = frag_a(Qs + wave * 16 * 256, 256, kc * 32, lane);
      v16h b = frag_b(Ks + wave * 16 * 256, 256, kc * 32, lane);
      e = wmma16(a, b, e);
    }
    int n = lane & 15, m0 = (lane >> 4) << 3;
#pragma unroll
    for (int r = 0; r < 8; ++r) {
      int m = m0 + r;
      bool ok = ((m >> 3) == (n >> 3)) && ((n & 7) <= (m & 7));
      ew[(wave * 16 + m) * 16 + n] = ok ? e[r] * 0.0625f : -9e15f;
    }
  }
  __syncthreads();

  // softmax: one row per thread (64 rows, 8 valid cols)
  if (tid < 64) {
    int w2i = tid >> 4, m = tid & 15;
    int nb = (m >> 3) << 3;
    float v[8], mx = -3.4e38f;
#pragma unroll
    for (int j = 0; j < 8; ++j) {
      v[j] = ew[(w2i * 16 + m) * 16 + nb + j];
      mx = fmaxf(mx, v[j]);
    }
    float s = 0.f;
#pragma unroll
    for (int j = 0; j < 8; ++j) { v[j] = __expf(v[j] - mx); s += v[j]; }
    float inv = 1.f / s;
#pragma unroll
    for (int j = 0; j < 8; ++j) attnS[tid * 8 + j] = (_Float16)(v[j] * inv);
  }
  __syncthreads();

  // out = attn @ V (8-deep)
  {
    int tr = tid & 63;
    int dh = tid >> 6;
    int vb = (tr >> 3) << 3;
    float aw[8];
#pragma unroll
    for (int j = 0; j < 8; ++j) aw[j] = (float)attnS[tr * 8 + j];
    for (int dc = 0; dc < 16; ++dc) {
      int d = dh * 128 + dc * 8;
      float acc[8] = {0,0,0,0,0,0,0,0};
#pragma unroll
      for (int s = 0; s < 8; ++s) {
        v8h vv = *(const v8h*)(Vs + (vb + s) * 256 + d);
#pragma unroll
        for (int j = 0; j < 8; ++j) acc[j] += aw[s] * (float)vv[j];
      }
      v8h o;
#pragma unroll
      for (int j = 0; j < 8; ++j) o[j] = (_Float16)acc[j];
      *(v8h*)(dstOuts + tr * 256 + d) = o;
    }
  }
}

// ---------------------------------------------------------------------------
// Stage 1: per-head self-attention + FFN.  Grid: (rowTiles, H), block 128.
// LDS: Xs 32K | Qs(outs) 32K | Ks 32K | Vs 32K | h1s 32K | slab 2x16K | ew 4K | attn 1K
// ---------------------------------------------------------------------------
__global__ __launch_bounds__(128) void ta9844_stage1(
    const _Float16* __restrict__ xh,
    const _Float16* __restrict__ WqT, const _Float16* __restrict__ WkT,
    const _Float16* __restrict__ WvT,
    const _Float16* __restrict__ W1T, const float* __restrict__ hb1,
    const _Float16* __restrict__ W2T, const float* __restrict__ hb2,
    _Float16* __restrict__ feats) {
  extern __shared__ char smem[];
  _Float16* Xs    = (_Float16*)(smem);
  _Float16* Qs    = (_Float16*)(smem + 32768);   // reused as "outs"
  _Float16* Ks    = (_Float16*)(smem + 65536);
  _Float16* Vs    = (_Float16*)(smem + 98304);
  _Float16* h1s   = (_Float16*)(smem + 131072);
  _Float16* slab  = (_Float16*)(smem + 163840);  // 2 x 16 KB
  float*    ew    = (float*)(smem + 196608);
  _Float16* attnS = (_Float16*)(smem + 200704);

  const int tid  = threadIdx.x;
  const int lane = tid & 31;
  const int wave = tid >> 5;
  const int h    = blockIdx.y;
  const long row0 = (long)blockIdx.x * 64;

  const _Float16* wq = WqT + (long)h * 256 * 256;
  const _Float16* wk = WkT + (long)h * 256 * 256;
  const _Float16* wv = WvT + (long)h * 256 * 256;
  const _Float16* w1 = W1T + (long)h * 256 * 512;
  const _Float16* w2 = W2T + (long)h * 256 * 256;
  const float* b1 = hb1 + h * 256;
  const float* b2 = hb2 + h * 256;

  // stage X tile (64 x 256 f16) into LDS
  for (int idx = tid; idx < 64 * 32; idx += 128) {
    int rr = idx >> 5;
    int cc = (idx & 31) << 3;
    *(v8h*)(Xs + rr * 256 + cc) = *(const v8h*)(xh + (row0 + rr) * 256 + cc);
  }

  const _Float16* xa = Xs + wave * 16 * 256;
  v8f acc[16];

  // --- Q, K, V projections (K=256), weight panels staged in LDS ---
#pragma unroll
  for (int i = 0; i < 16; ++i) acc[i] = zero8();
  gemm256<8>(wq, 256, xa, 256, 8, xa, 256, slab, acc, tid, lane);
#pragma unroll
  for (int nt = 0; nt < 16; ++nt)
    store_tile_h(Qs + wave * 16 * 256 + nt * 16, 256, acc[nt], lane);

#pragma unroll
  for (int i = 0; i < 16; ++i) acc[i] = zero8();
  gemm256<8>(wk, 256, xa, 256, 8, xa, 256, slab, acc, tid, lane);
#pragma unroll
  for (int nt = 0; nt < 16; ++nt)
    store_tile_h(Ks + wave * 16 * 256 + nt * 16, 256, acc[nt], lane);

#pragma unroll
  for (int i = 0; i < 16; ++i) acc[i] = zero8();
  gemm256<8>(wv, 256, xa, 256, 8, xa, 256, slab, acc, tid, lane);
#pragma unroll
  for (int nt = 0; nt < 16; ++nt)
    store_tile_h(Vs + wave * 16 * 256 + nt * 16, 256, acc[nt], lane);
  __syncthreads();

  // --- attention (writes outs == Qs) ---
  attention_block(Qs, Ks, Vs, Qs, ew, attnS, tid, lane, wave);

  // --- FFN1: relu(concat[out, x] @ W1 + b1), K = 512 ---
#pragma unroll
  for (int i = 0; i < 16; ++i) acc[i] = zero8();
  gemm256<16>(w1, 512, Qs + wave * 16 * 256, 256, 8, xa, 256, slab, acc, tid, lane);
#pragma unroll
  for (int nt = 0; nt < 16; ++nt) {
    float bias = b1[nt * 16 + (lane & 15)];
    v8f a = acc[nt];
#pragma unroll
    for (int r = 0; r < 8; ++r) a[r] = fmaxf(a[r] + bias, 0.f);
    store_tile_h(h1s + wave * 16 * 256 + nt * 16, 256, a, lane);
  }

  // --- FFN2: h1 @ W2 + b2 -> feats (f16) ---
#pragma unroll
  for (int i = 0; i < 16; ++i) acc[i] = zero8();
  gemm256<8>(w2, 256, h1s + wave * 16 * 256, 256, 8,
             h1s + wave * 16 * 256, 256, slab, acc, tid, lane);
#pragma unroll
  for (int nt = 0; nt < 16; ++nt) {
    float bias = b2[nt * 16 + (lane & 15)];
    int n = lane & 15, m0 = (lane >> 4) << 3;
#pragma unroll
    for (int r = 0; r < 8; ++r) {
      long row = row0 + wave * 16 + m0 + r;
      feats[row * 2048 + h * 256 + nt * 16 + n] = (_Float16)(acc[nt][r] + bias);
    }
  }
}

// ---------------------------------------------------------------------------
// Stage 2: output self-attention over concatenated head features (Din=2048).
// LDS: Qs 32K | Ks 32K | Vs 32K | outs 32K | h1s 32K | slab 2x16K | ew 4K | attn 1K
// ---------------------------------------------------------------------------
__global__ __launch_bounds__(128) void ta9844_stage2(
    const _Float16* __restrict__ feats,
    const _Float16* __restrict__ WqT, const _Float16* __restrict__ WkT,
    const _Float16* __restrict__ WvT,
    const _Float16* __restrict__ W1T, const float* __restrict__ b1,
    const _Float16* __restrict__ W2T, const float* __restrict__ b2,
    float* __restrict__ out) {
  extern __shared__ char smem[];
  _Float16* Qs    = (_Float16*)(smem);
  _Float16* Ks    = (_Float16*)(smem + 32768);
  _Float16* Vs    = (_Float16*)(smem + 65536);
  _Float16* outs  = (_Float16*)(smem + 98304);
  _Float16* h1s   = (_Float16*)(smem + 131072);
  _Float16* slab  = (_Float16*)(smem + 163840);  // 2 x 16 KB
  float*    ew    = (float*)(smem + 196608);
  _Float16* attnS = (_Float16*)(smem + 200704);

  const int tid  = threadIdx.x;
  const int lane = tid & 31;
  const int wave = tid >> 5;
  const long row0 = (long)blockIdx.x * 64;
  const _Float16* fa = feats + (row0 + wave * 16) * 2048;

  v8f acc[16];

  // --- Q, K, V projections (K=2048), A streamed from global feats ---
#pragma unroll
  for (int i = 0; i < 16; ++i) acc[i] = zero8();
  gemm256<64>(WqT, 2048, fa, 2048, 64, fa, 2048, slab, acc, tid, lane);
#pragma unroll
  for (int nt = 0; nt < 16; ++nt)
    store_tile_h(Qs + wave * 16 * 256 + nt * 16, 256, acc[nt], lane);

#pragma unroll
  for (int i = 0; i < 16; ++i) acc[i] = zero8();
  gemm256<64>(WkT, 2048, fa, 2048, 64, fa, 2048, slab, acc, tid, lane);
#pragma unroll
  for (int nt = 0; nt < 16; ++nt)
    store_tile_h(Ks + wave * 16 * 256 + nt * 16, 256, acc[nt], lane);

#pragma unroll
  for (int i = 0; i < 16; ++i) acc[i] = zero8();
  gemm256<64>(WvT, 2048, fa, 2048, 64, fa, 2048, slab, acc, tid, lane);
#pragma unroll
  for (int nt = 0; nt < 16; ++nt)
    store_tile_h(Vs + wave * 16 * 256 + nt * 16, 256, acc[nt], lane);
  __syncthreads();

  // --- attention ---
  attention_block(Qs, Ks, Vs, outs, ew, attnS, tid, lane, wave);

  // --- FFN1: relu(concat[out(256), feats(2048)] @ W1 + b1), K = 2304 ---
#pragma unroll
  for (int i = 0; i < 16; ++i) acc[i] = zero8();
  gemm256<72>(W1T, 2304, outs + wave * 16 * 256, 256, 8, fa, 2048,
              slab, acc, tid, lane);
#pragma unroll
  for (int nt = 0; nt < 16; ++nt) {
    float bias = b1[nt * 16 + (lane & 15)];
    v8f a = acc[nt];
#pragma unroll
    for (int r = 0; r < 8; ++r) a[r] = fmaxf(a[r] + bias, 0.f);
    store_tile_h(h1s + wave * 16 * 256 + nt * 16, 256, a, lane);
  }

  // --- FFN2: h1 @ W2 + b2 -> final f32 output ---
#pragma unroll
  for (int i = 0; i < 16; ++i) acc[i] = zero8();
  gemm256<8>(W2T, 256, h1s + wave * 16 * 256, 256, 8,
             h1s + wave * 16 * 256, 256, slab, acc, tid, lane);
#pragma unroll
  for (int nt = 0; nt < 16; ++nt) {
    float bias = b2[nt * 16 + (lane & 15)];
    int n = lane & 15, m0 = (lane >> 4) << 3;
#pragma unroll
    for (int r = 0; r < 8; ++r) {
      long row = row0 + wave * 16 + m0 + r;
      out[row * 256 + nt * 16 + n] = acc[nt][r] + bias;
    }
  }
}

// ---------------------------------------------------------------------------
// Launch
// ---------------------------------------------------------------------------
extern "C" void kernel_launch(void* const* d_in, const int* in_sizes, int n_in,
                              void* d_out, int out_size, void* d_ws, size_t ws_size,
                              hipStream_t stream) {
  const float* x   = (const float*)d_in[0];
  const float* pos = (const float*)d_in[1];
  const float* hWq = (const float*)d_in[2];
  const float* hWk = (const float*)d_in[3];
  const float* hWv = (const float*)d_in[4];
  const float* hW1 = (const float*)d_in[5];
  const float* hb1 = (const float*)d_in[6];
  const float* hW2 = (const float*)d_in[7];
  const float* hb2 = (const float*)d_in[8];
  const float* oWq = (const float*)d_in[9];
  const float* oWk = (const float*)d_in[10];
  const float* oWv = (const float*)d_in[11];
  const float* oW1 = (const float*)d_in[12];
  const float* ob1 = (const float*)d_in[13];
  const float* oW2 = (const float*)d_in[14];
  const float* ob2 = (const float*)d_in[15];

  const long rows  = (long)in_sizes[0] / 256;    // B*T = 131072
  const int  H     = 8;
  const int  tiles = (int)(rows / 64);           // 2048

  // Workspace carve-up (f16 buffers)
  char* ws = (char*)d_ws;
  size_t off = 0;
  _Float16* xh    = (_Float16*)(ws + off); off += (size_t)rows * 256 * 2;
  _Float16* feats = (_Float16*)(ws + off); off += (size_t)rows * 2048 * 2;
  _Float16* WqT   = (_Float16*)(ws + off); off += (size_t)H * 256 * 256 * 2;
  _Float16* WkT   = (_Float16*)(ws + off); off += (size_t)H * 256 * 256 * 2;
  _Float16* WvT   = (_Float16*)(ws + off); off += (size_t)H * 256 * 256 * 2;
  _Float16* W1T   = (_Float16*)(ws + off); off += (size_t)H * 256 * 512 * 2;
  _Float16* W2T   = (_Float16*)(ws + off); off += (size_t)H * 256 * 256 * 2;
  _Float16* oWqT  = (_Float16*)(ws + off); off += (size_t)256 * 2048 * 2;
  _Float16* oWkT  = (_Float16*)(ws + off); off += (size_t)256 * 2048 * 2;
  _Float16* oWvT  = (_Float16*)(ws + off); off += (size_t)256 * 2048 * 2;
  _Float16* oW1T  = (_Float16*)(ws + off); off += (size_t)256 * 2304 * 2;
  _Float16* oW2T  = (_Float16*)(ws + off); off += (size_t)256 * 256 * 2;

  {
    long total = rows * 256;
    ta9844_prep_x<<<(int)((total + 255) / 256), 256, 0, stream>>>(x, pos, xh, total);
  }
  auto wT = [&](const float* s, _Float16* d, int M, int K, int N) {
    long tot = (long)M * K * N;
    ta9844_prep_wT<<<(int)((tot + 255) / 256), 256, 0, stream>>>(s, d, M, K, N);
  };
  wT(hWq, WqT, H, 256, 256);
  wT(hWk, WkT, H, 256, 256);
  wT(hWv, WvT, H, 256, 256);
  wT(hW1, W1T, H, 512, 256);
  wT(hW2, W2T, H, 256, 256);
  wT(oWq, oWqT, 1, 2048, 256);
  wT(oWk, oWkT, 1, 2048, 256);
  wT(oWv, oWvT, 1, 2048, 256);
  wT(oW1, oW1T, 1, 2304, 256);
  wT(oW2, oW2T, 1, 256, 256);

  const int LDS_BYTES = 201728;  // < 320 KB/WGP

  ta9844_stage1<<<dim3(tiles, H), 128, LDS_BYTES, stream>>>(
      xh, WqT, WkT, WvT, W1T, hb1, W2T, hb2, feats);

  ta9844_stage2<<<dim3(tiles), 128, LDS_BYTES, stream>>>(
      feats, oWqT, oWkT, oWvT, oW1T, ob1, oW2T, ob2, (float*)d_out);
}
